// FastWeightAttention_72662256714018
// MI455X (gfx1250) — matrix operand order repaired and checked by reading, not compiled, not run on hardware
//
#include <hip/hip_runtime.h>

// ---------------------------------------------------------------------------
// Multi-head attention for MI455X (gfx1250, wave32, WMMA bf16 f32-accumulate).
// Stages: prep (fp32->bf16 + transposes) -> QKV GEMM (double-buffered) ->
// flash attention (async-to-LDS K/V staging, online softmax) -> output GEMM.
// ---------------------------------------------------------------------------

typedef __bf16 bf16_t;
typedef __attribute__((ext_vector_type(16))) __bf16 v16bf;
typedef __attribute__((ext_vector_type(8)))  __bf16 v8bf;
typedef __attribute__((ext_vector_type(8)))  float  v8f;

#define DIM     1024
#define DH      64
#define HEADS   16
#define BATCH   2
#define SEQ     2048
#define SCALE   0.125f          // 64^-0.5
#define LOG2E   1.44269504f

// ---------------------------------------------------------------------------
// Fragment helpers (CDNA5 WMMA 16x16x32 bf16 layouts, wave32).
// A (16x32, MxK): lane L holds row m = L&15; half = L>>4.
//   element j<8  -> k = ks + 8*half + j        (contiguous 16B)
//   element j>=8 -> k = ks + 16 + 8*half + j-8 (contiguous 16B)
// B (32x16, KxN), stored transposed BT[n][k] row-major: lane L holds column
//   n = L&15; element j -> same k mapping as A (two contiguous 16B runs).
// C/D (16x16 f32): element r -> (m = r + 8*half, n = lane&15).
// ---------------------------------------------------------------------------

__device__ __forceinline__ v16bf make_frag(v8bf lo, v8bf hi) {
  v16bf r;
#pragma unroll
  for (int i = 0; i < 8; ++i) { r[i] = lo[i]; r[i + 8] = hi[i]; }
  return r;
}

__device__ __forceinline__ v16bf load_a_frag(const bf16_t* __restrict__ src,
                                             int ld, int m, int ks, int half) {
  const bf16_t* p = src + (size_t)m * ld + ks;
  v8bf lo = *(const v8bf*)(p + 8 * half);
  v8bf hi = *(const v8bf*)(p + 16 + 8 * half);
  return make_frag(lo, hi);
}

__device__ __forceinline__ v16bf load_bt_frag(const bf16_t* __restrict__ bt,
                                              int ld, int n, int ks, int half) {
  const bf16_t* p = bt + (size_t)n * ld + ks;
  v8bf lo = *(const v8bf*)(p + 8 * half);
  v8bf hi = *(const v8bf*)(p + 16 + 8 * half);
  return make_frag(lo, hi);
}

__device__ __forceinline__ v8f wmma_bf16(v16bf a, v16bf b, v8f c) {
  return __builtin_amdgcn_wmma_f32_16x16x32_bf16(
      /*neg_a=*/false, a, /*neg_b=*/false, b,
      /*c_mod=*/(short)0, c, /*reuse_a=*/false, /*reuse_b=*/false);
}

__device__ __forceinline__ void zero8(v8f& v) {
#pragma unroll
  for (int j = 0; j < 8; ++j) v[j] = 0.0f;
}

// CDNA5 async DMA: global -> LDS, 128b per lane, tracked by ASYNCcnt.
__device__ __forceinline__ void async_b128(unsigned lds_off, const void* g) {
  asm volatile("global_load_async_to_lds_b128 %0, %1, off"
               :: "v"(lds_off), "v"((unsigned long long)(uintptr_t)g)
               : "memory");
}
#define WAIT_ASYNC(n) asm volatile("s_wait_asynccnt " #n ::: "memory")
#define WAIT_DS0()    asm volatile("s_wait_dscnt 0" ::: "memory")

// ---------------------------------------------------------------------------
// Stage 0: precision / layout prep
// ---------------------------------------------------------------------------

__global__ __launch_bounds__(256) void k_prep_tokens(
    const float* __restrict__ t, bf16_t* __restrict__ o, int n) {
  for (int i = blockIdx.x * blockDim.x + threadIdx.x; i < n;
       i += gridDim.x * blockDim.x)
    o[i] = (bf16_t)t[i];
}

// w[h][d][e] (fp32) -> wt[h][e][d] (bf16), h<16, d<1024, e<64
__global__ __launch_bounds__(256) void k_prep_wT(
    const float* __restrict__ w, bf16_t* __restrict__ wt) {
  int i = blockIdx.x * blockDim.x + threadIdx.x;   // over 16*64*1024
  int h = i >> 16;
  int r = i & 65535;
  int e = r >> 10;
  int d = r & 1023;
  wt[i] = (bf16_t)w[((size_t)h * DIM + d) * DH + e];
}

// wo[h][e][d] (fp32) -> woT[d][h*64+e] (bf16)
__global__ __launch_bounds__(256) void k_prep_woT(
    const float* __restrict__ wo, bf16_t* __restrict__ wot) {
  int i = blockIdx.x * blockDim.x + threadIdx.x;   // over 1024*1024 (out idx)
  int d = i >> 10;
  int c = i & 1023;                                 // h*64+e
  wot[i] = (bf16_t)wo[(size_t)c * DIM + d];
}

// ---------------------------------------------------------------------------
// Stage 1: QKV projection.  A = tokens_bf [4096][1024], B = wT[h] [64][1024].
// blockIdx.y = h*3 + which (0=Q scaled, 1=K, 2=V stored transposed).
// 8 waves/block, one 16x64 output tile per wave; k-loop software-pipelined.
// ---------------------------------------------------------------------------

__global__ __launch_bounds__(256) void k_qkv(
    const bf16_t* __restrict__ tok,
    const bf16_t* __restrict__ wqT, const bf16_t* __restrict__ wkT,
    const bf16_t* __restrict__ wvT,
    bf16_t* __restrict__ Q, bf16_t* __restrict__ K, bf16_t* __restrict__ Vt) {
  const int wave = threadIdx.x >> 5;
  const int lane = threadIdx.x & 31;
  const int half = lane >> 4, lc = lane & 15;
  const int which = blockIdx.y % 3;
  const int h     = blockIdx.y / 3;
  const int m0 = (blockIdx.x * 8 + wave) * 16;      // row in [0,4096)

  const bf16_t* WT = (which == 0 ? wqT : (which == 1 ? wkT : wvT)) +
                     (size_t)h * DH * DIM;
  const bf16_t* Arow = tok + (size_t)m0 * DIM;

  v8f acc[4];
#pragma unroll
  for (int i = 0; i < 4; ++i) zero8(acc[i]);

  // Double-buffered k-loop: fragments for step s+1 load while step s computes.
  v16bf a0 = load_a_frag(Arow, DIM, lc, 0, half);
  v16bf b0[4], b1[4];
#pragma unroll
  for (int nt = 0; nt < 4; ++nt)
    b0[nt] = load_bt_frag(WT, DIM, nt * 16 + lc, 0, half);

  for (int ks = 0; ks < DIM; ks += 64) {
    v16bf a1 = load_a_frag(Arow, DIM, lc, ks + 32, half);
#pragma unroll
    for (int nt = 0; nt < 4; ++nt)
      b1[nt] = load_bt_frag(WT, DIM, nt * 16 + lc, ks + 32, half);
#pragma unroll
    for (int nt = 0; nt < 4; ++nt) acc[nt] = wmma_bf16(a0, b0[nt], acc[nt]);
    if (ks + 64 < DIM) {
      a0 = load_a_frag(Arow, DIM, lc, ks + 64, half);
#pragma unroll
      for (int nt = 0; nt < 4; ++nt)
        b0[nt] = load_bt_frag(WT, DIM, nt * 16 + lc, ks + 64, half);
    }
#pragma unroll
    for (int nt = 0; nt < 4; ++nt) acc[nt] = wmma_bf16(a1, b1[nt], acc[nt]);
  }

#pragma unroll
  for (int nt = 0; nt < 4; ++nt)
#pragma unroll
    for (int r = 0; r < 8; ++r) {
      int row = m0 + r + 8 * half;                  // global (b,s) row
      int b   = row >> 11;
      int s   = row & (SEQ - 1);
      int e   = nt * 16 + lc;
      float v = acc[nt][r];
      size_t bh = (size_t)b * HEADS + h;
      if (which == 0)
        Q[(bh * SEQ + s) * DH + e] = (bf16_t)(v * SCALE);
      else if (which == 1)
        K[(bh * SEQ + s) * DH + e] = (bf16_t)v;
      else
        Vt[(bh * DH + e) * SEQ + s] = (bf16_t)v;
    }
}

// ---------------------------------------------------------------------------
// Stage 2: flash attention.  Block = 8 waves x 16 query rows = 128 queries.
// K/Vt tiles (64 kv) DMA'd to double-buffered LDS with async-to-LDS loads
// (ASYNCcnt); next tile's DMA overlaps current tile's WMMA + softmax.
// P round-trips through a per-wave LDS tile (C-layout -> A-layout),
// guarded by s_wait_dscnt (same-wave LDS ordering).
// ---------------------------------------------------------------------------

// Issue one KV tile (K: 8KB contiguous, Vt: 64 rows x 128B) as async DMA.
// 4 async instructions per wave (2 per tile); 512 16B chunks / 256 threads.
__device__ __forceinline__ void issue_tile(const bf16_t* __restrict__ Kg,
                                           const bf16_t* __restrict__ Vtg,
                                           int kv, unsigned ksOff,
                                           unsigned vsOff, int tid) {
  const char* kga = (const char*)(Kg + (size_t)kv * DH);
#pragma unroll
  for (int u = 0; u < 2; ++u) {
    int c = tid + u * 256;
    async_b128(ksOff + c * 16, kga + c * 16);
  }
#pragma unroll
  for (int u = 0; u < 2; ++u) {
    int c = tid + u * 256;
    int d = c >> 3, j8 = (c & 7) * 8;
    async_b128(vsOff + c * 16,
               (const char*)(Vtg + (size_t)d * SEQ + kv + j8));
  }
}

__global__ __launch_bounds__(256) void k_flash(
    const bf16_t* __restrict__ Q, const bf16_t* __restrict__ K,
    const bf16_t* __restrict__ Vt, bf16_t* __restrict__ agg) {
  __shared__ alignas(16) bf16_t Ks[2][64 * 64];   // K tiles  [n][k]  2x8 KB
  __shared__ alignas(16) bf16_t Vs[2][64 * 64];   // Vt tiles [d][j]  2x8 KB
  __shared__ alignas(16) bf16_t Ps[8 * 16 * 64];  // per-wave P tiles 16 KB

  const int tid  = threadIdx.x;
  const int wave = tid >> 5;
  const int lane = tid & 31;
  const int half = lane >> 4, lc = lane & 15;
  const int b = blockIdx.y >> 4;
  const int h = blockIdx.y & 15;
  const int m0 = blockIdx.x * 128 + wave * 16;      // query row in [0,2048)

  const size_t bh = (size_t)b * HEADS + h;
  const bf16_t* Qbh  = Q  + bh * SEQ * DH;
  const bf16_t* Kbh  = K  + bh * SEQ * DH;
  const bf16_t* Vtbh = Vt + bh * DH * SEQ;

  const unsigned ksOff0 = (unsigned)(uintptr_t)(&Ks[0][0]);
  const unsigned ksOff1 = (unsigned)(uintptr_t)(&Ks[1][0]);
  const unsigned vsOff0 = (unsigned)(uintptr_t)(&Vs[0][0]);
  const unsigned vsOff1 = (unsigned)(uintptr_t)(&Vs[1][0]);

  // Q fragments held in registers for the whole pass (dh=64 -> 2 k-steps)
  v16bf qf[2];
  qf[0] = load_a_frag(Qbh + (size_t)m0 * DH, DH, lc, 0,  half);
  qf[1] = load_a_frag(Qbh + (size_t)m0 * DH, DH, lc, 32, half);

  v8f O[4];
#pragma unroll
  for (int i = 0; i < 4; ++i) zero8(O[i]);
  float mrow[8], lrow[8];
#pragma unroll
  for (int r = 0; r < 8; ++r) { mrow[r] = -3.0e38f; lrow[r] = 0.0f; }

  bf16_t* Pw = Ps + wave * 16 * 64;

  // Prologue: DMA tile 0 into buffer 0.
  issue_tile(Kbh, Vtbh, 0, ksOff0, vsOff0, tid);

  for (int kv = 0; kv < SEQ; kv += 64) {
    const int pp = (kv >> 6) & 1;
    const bf16_t* KsC = Ks[pp];
    const bf16_t* VsC = Vs[pp];

    // DMA next tile into the other buffer (safe: barrier at end of previous
    // iteration guarantees nobody is still reading it), then wait for the
    // current tile (async loads complete in order; 4 newly issued remain).
    if (kv + 64 < SEQ) {
      issue_tile(Kbh, Vtbh, kv + 64, pp ? ksOff0 : ksOff1,
                 pp ? vsOff0 : vsOff1, tid);
      WAIT_ASYNC(4);
    } else {
      WAIT_ASYNC(0);
    }
    __syncthreads();                      // current tile visible to all waves

    // ---- sim = Q K^T (Q already scaled) : 16x64 tile per wave -------------
    v8f s[4];
#pragma unroll
    for (int i = 0; i < 4; ++i) zero8(s[i]);
#pragma unroll
    for (int ks2 = 0; ks2 < 2; ++ks2)
#pragma unroll
      for (int nt = 0; nt < 4; ++nt) {
        v16bf bb = load_bt_frag(KsC, 64, nt * 16 + lc, ks2 * 32, half);
        s[nt] = wmma_bf16(qf[ks2], bb, s[nt]);
      }

    // ---- online softmax (rows live across lanes 0..15 within each half) ---
    float mnew[8], corr[8];
#pragma unroll
    for (int r = 0; r < 8; ++r) {
      float mx = fmaxf(fmaxf(s[0][r], s[1][r]), fmaxf(s[2][r], s[3][r]));
      mx = fmaxf(mx, __shfl_xor(mx, 1, 32));
      mx = fmaxf(mx, __shfl_xor(mx, 2, 32));
      mx = fmaxf(mx, __shfl_xor(mx, 4, 32));
      mx = fmaxf(mx, __shfl_xor(mx, 8, 32));
      mnew[r] = fmaxf(mrow[r], mx);
      corr[r] = exp2f((mrow[r] - mnew[r]) * LOG2E);
      mrow[r] = mnew[r];
    }
    float rsum[8];
#pragma unroll
    for (int r = 0; r < 8; ++r) rsum[r] = 0.0f;
#pragma unroll
    for (int nt = 0; nt < 4; ++nt)
#pragma unroll
      for (int r = 0; r < 8; ++r) {
        float p = exp2f((s[nt][r] - mnew[r]) * LOG2E);
        s[nt][r] = p;
        rsum[r] += p;
      }
#pragma unroll
    for (int r = 0; r < 8; ++r) {
      float t = rsum[r];
      t += __shfl_xor(t, 1, 32);
      t += __shfl_xor(t, 2, 32);
      t += __shfl_xor(t, 4, 32);
      t += __shfl_xor(t, 8, 32);
      lrow[r] = lrow[r] * corr[r] + t;
#pragma unroll
      for (int nt = 0; nt < 4; ++nt) O[nt][r] *= corr[r];
    }

    // ---- P: C-layout -> row-major LDS -> A-layout reload ------------------
#pragma unroll
    for (int nt = 0; nt < 4; ++nt)
#pragma unroll
      for (int r = 0; r < 8; ++r)
        Pw[(r + 8 * half) * 64 + nt * 16 + lc] = (bf16_t)s[nt][r];
    WAIT_DS0();                           // same-wave LDS RAW

    // ---- O += P V ---------------------------------------------------------
#pragma unroll
    for (int js = 0; js < 2; ++js) {
      v16bf pa = load_a_frag(Pw, 64, lc, js * 32, half);
#pragma unroll
      for (int dt = 0; dt < 4; ++dt) {
        v16bf vb = load_bt_frag(VsC, 64, dt * 16 + lc, js * 32, half);
        O[dt] = wmma_bf16(pa, vb, O[dt]);
      }
    }
    __syncthreads();                      // done reading cur before reuse
  }

  // ---- epilogue: normalize, store agg[b][s][h*64+d] as bf16 ---------------
#pragma unroll
  for (int nt = 0; nt < 4; ++nt)
#pragma unroll
    for (int r = 0; r < 8; ++r) {
      int s_idx = m0 + r + 8 * half;
      int d     = nt * 16 + lc;
      float v   = O[nt][r] / lrow[r];
      agg[((size_t)b * SEQ + s_idx) * DIM + h * DH + d] = (bf16_t)v;
    }
}

// ---------------------------------------------------------------------------
// Stage 3: output projection.  [4096,1024] x [1024,1024] -> fp32 d_out.
// Same double-buffered k-loop as stage 1.
// ---------------------------------------------------------------------------

__global__ __launch_bounds__(256) void k_oproj(
    const bf16_t* __restrict__ agg, const bf16_t* __restrict__ woT,
    float* __restrict__ out) {
  const int wave = threadIdx.x >> 5;
  const int lane = threadIdx.x & 31;
  const int half = lane >> 4, lc = lane & 15;
  const int m0 = (blockIdx.x * 8 + wave) * 16;      // [0,4096)
  const int n0 = blockIdx.y * 64;                   // [0,1024)
  const bf16_t* Arow = agg + (size_t)m0 * DIM;

  v8f acc[4];
#pragma unroll
  for (int i = 0; i < 4; ++i) zero8(acc[i]);

  v16bf a0 = load_a_frag(Arow, DIM, lc, 0, half);
  v16bf b0[4], b1[4];
#pragma unroll
  for (int nt = 0; nt < 4; ++nt)
    b0[nt] = load_bt_frag(woT, DIM, n0 + nt * 16 + lc, 0, half);

  for (int ks = 0; ks < DIM; ks += 64) {
    v16bf a1 = load_a_frag(Arow, DIM, lc, ks + 32, half);
#pragma unroll
    for (int nt = 0; nt < 4; ++nt)
      b1[nt] = load_bt_frag(woT, DIM, n0 + nt * 16 + lc, ks + 32, half);
#pragma unroll
    for (int nt = 0; nt < 4; ++nt) acc[nt] = wmma_bf16(a0, b0[nt], acc[nt]);
    if (ks + 64 < DIM) {
      a0 = load_a_frag(Arow, DIM, lc, ks + 64, half);
#pragma unroll
      for (int nt = 0; nt < 4; ++nt)
        b0[nt] = load_bt_frag(woT, DIM, n0 + nt * 16 + lc, ks + 64, half);
    }
#pragma unroll
    for (int nt = 0; nt < 4; ++nt) acc[nt] = wmma_bf16(a1, b1[nt], acc[nt]);
  }

#pragma unroll
  for (int nt = 0; nt < 4; ++nt)
#pragma unroll
    for (int r = 0; r < 8; ++r)
      out[(size_t)(m0 + r + 8 * half) * DIM + n0 + nt * 16 + lc] = acc[nt][r];
}

// ---------------------------------------------------------------------------
// Launcher.  Workspace layout (bytes):
//   tok_bf @0 (8MB) | wqT @8M | wkT @10M | wvT @12M | woT @14M (2MB each)
//   Q @16M | K @24M | Vt @32M (8MB each, bf16) | agg @40M (8MB bf16)
// ---------------------------------------------------------------------------

extern "C" void kernel_launch(void* const* d_in, const int* in_sizes, int n_in,
                              void* d_out, int out_size, void* d_ws,
                              size_t ws_size, hipStream_t stream) {
  const float* tokens = (const float*)d_in[0];
  const float* wq     = (const float*)d_in[1];
  const float* wk     = (const float*)d_in[2];
  const float* wv     = (const float*)d_in[3];
  const float* wo     = (const float*)d_in[4];
  float* out          = (float*)d_out;

  char* ws = (char*)d_ws;
  const size_t MB = 1024ull * 1024ull;
  bf16_t* tok_bf = (bf16_t*)(ws + 0 * MB);
  bf16_t* wqT    = (bf16_t*)(ws + 8 * MB);
  bf16_t* wkT    = (bf16_t*)(ws + 10 * MB);
  bf16_t* wvT    = (bf16_t*)(ws + 12 * MB);
  bf16_t* woT    = (bf16_t*)(ws + 14 * MB);
  bf16_t* Qb     = (bf16_t*)(ws + 16 * MB);
  bf16_t* Kb     = (bf16_t*)(ws + 24 * MB);
  bf16_t* Vtb    = (bf16_t*)(ws + 32 * MB);
  bf16_t* aggb   = (bf16_t*)(ws + 40 * MB);

  // Stage 0: conversions / transposes
  k_prep_tokens<<<2048, 256, 0, stream>>>(tokens, tok_bf, BATCH * SEQ * DIM);
  k_prep_wT<<<4096, 256, 0, stream>>>(wq, wqT);
  k_prep_wT<<<4096, 256, 0, stream>>>(wk, wkT);
  k_prep_wT<<<4096, 256, 0, stream>>>(wv, wvT);
  k_prep_woT<<<4096, 256, 0, stream>>>(wo, woT);

  // Stage 1: QKV projection (grid.y = head*3 + {q,k,v})
  k_qkv<<<dim3(32, HEADS * 3), 256, 0, stream>>>(tok_bf, wqT, wkT, wvT,
                                                 Qb, Kb, Vtb);

  // Stage 2: flash attention (grid = 16 query-tiles x (b*16+h))
  k_flash<<<dim3(SEQ / 128, BATCH * HEADS), 256, 0, stream>>>(Qb, Kb, Vtb,
                                                              aggb);

  // Stage 3: output projection
  k_oproj<<<dim3(32, DIM / 64), 256, 0, stream>>>(aggb, woT, out);
}
